// RecurrentQ_64914135712063
// MI455X (gfx1250) — matrix operand order repaired and checked by reading, not compiled, not run on hardware
//
#include <hip/hip_runtime.h>
#include <hip/hip_bf16.h>
#include <math.h>

// ---------------------------------------------------------------------------
// LSTM rollout for MI455X (gfx1250), bf16 WMMA (v_wmma_f32_16x16x32_bf16).
//
//  k_init      : fp32 -> bf16 conversion of weights/state, zero h/c/barrier
//  k_encoder   : encoded = relu(state @ W_enc^T + b_enc)            (WMMA)
//  k_recurrent : persistent 32-block kernel, T=256 steps, grid barrier per
//                step. Each wave: M=32 (2 row-tiles) x 4 gate tiles, so every
//                weight (B) fragment is reused twice -> W L2 traffic 64MB/step.
//                enc_t / h row slabs staged into LDS via the Tensor Data
//                Mover (tensor_load_to_lds + s_wait_tensorcnt), fragments
//                double-buffered in registers to break load->wmma waits.
//  k_decoder   : values = (latent @ W_dec^T + b_dec) * mask         (WMMA)
//  k_finalize  : copy final h, c into d_out
// ---------------------------------------------------------------------------

#define T_STEPS 256
#define BATCH   128
#define OBSD    512
#define HID     1024
#define ACT     18
#define NPAD    32          // decoder N padded to 32 cols (2 WMMA tiles)

typedef __bf16 bf16;
typedef __attribute__((ext_vector_type(16))) __bf16 bf16x16;
typedef __attribute__((ext_vector_type(8)))  __bf16 bf16x8;
typedef __attribute__((ext_vector_type(8)))  float  f32x8;
typedef __attribute__((ext_vector_type(4)))  unsigned int u32x4;
typedef __attribute__((ext_vector_type(8)))  int i32x8;
typedef __attribute__((ext_vector_type(4)))  int i32x4;

// ---- WMMA wrapper ----------------------------------------------------------
__device__ __forceinline__ f32x8 wmma_bf16(bf16x16 a, bf16x16 b, f32x8 c) {
    // 8 args: (neg_a, A, neg_b, B, c_mod, C, reuse_a, reuse_b)
    return __builtin_amdgcn_wmma_f32_16x16x32_bf16(false, a, false, b,
                                                   (short)0, c, false, false);
}

// ---- A fragment: 16x32 bf16 tile, row-major source, base at (row0, k0).
// ISA 7.12.2: lanes 0-15 -> M=lane, K 0..7 & 16..23; lanes 16-31 -> K 8..15 & 24..31
__device__ __forceinline__ bf16x16 ldfragA(const bf16* base, int stride) {
    const int lane = threadIdx.x & 31;
    const bf16* p = base + (size_t)(lane & 15) * stride + ((lane >> 4) << 3);
    bf16x8 lo = *(const bf16x8*)(p);        // K +0..7   (or +8..15)
    bf16x8 hi = *(const bf16x8*)(p + 16);   // K +16..23 (or +24..31)
    bf16x16 f;
#pragma unroll
    for (int i = 0; i < 8; ++i) { f[i] = lo[i]; f[i + 8] = hi[i]; }
    return f;
}

// ---- B fragment: 32x16 bf16 = W^T tile; W row-major (N rows x K cols),
// base at (n0, k0). Lanes 0-15 -> col=lane, K 0..7 & 16..23; lanes 16-31 -> K 8..15 & 24..31.
__device__ __forceinline__ bf16x16 ldfragB(const bf16* base, int stride) {
    const int lane = threadIdx.x & 31;
    const bf16* p = base + (size_t)(lane & 15) * stride + ((lane >> 4) << 3);
    bf16x8 lo = *(const bf16x8*)(p);
    bf16x8 hi = *(const bf16x8*)(p + 16);
    bf16x16 f;
#pragma unroll
    for (int i = 0; i < 8; ++i) { f[i] = lo[i]; f[i + 8] = hi[i]; }
    return f;
}

__device__ __forceinline__ float sigmoidf_(float x) {
    return 1.0f / (1.0f + __expf(-x));
}

// ---- Tensor Data Mover: DMA a 2D bf16 tile (rows x cols, row stride = cols,
// i.e. contiguous) from global memory into LDS.  D# layout per
// CDNA5 ISA (group0: count/lds/global/type,
// group1: data_size + tensor/tile dims + dim0 stride).
// This toolchain's builtin takes 6 args:
//   (uint32x4 g0, int32x8 g1, int32x4 g2, int32x4 g3, int32x8 pad, i32 cpol)
#if __has_builtin(__builtin_amdgcn_tensor_load_to_lds)
#define HAVE_TDM 1
__device__ __forceinline__ void tdm_load_2d(const void* gptr, void* lptr,
                                            unsigned rows, unsigned cols) {
    union { unsigned long long q[2]; u32x4 v; } g0;
    union { unsigned long long q[4]; i32x8 v; } g1;
    const unsigned long long ga  = (unsigned long long)(uintptr_t)gptr;
    const unsigned           lds = (unsigned)(uintptr_t)lptr; // LDS offset = addr[31:0]
    g0.q[0] = 1ull                                   // count = 1 (valid descriptor)
            | ((unsigned long long)lds << 32);       // lds_addr [63:32]
    g0.q[1] = (ga & 0x01FFFFFFFFFFFFFFull)           // global_addr [120:64] -> bits 56:0
            | (2ull << 62);                          // type = 2 ("image") [127:126]
    g1.q[0] = (1ull << 16)                           // data_size = 1 -> 2 bytes [17:16]
            | ((unsigned long long)(cols & 0xFFFFu) << 48);   // tensor_dim0[15:0] @bit48
    g1.q[1] = (unsigned long long)(cols >> 16)                // tensor_dim0[31:16] @bit64
            | ((unsigned long long)rows << 16)                // tensor_dim1 @bit80
            | ((unsigned long long)(cols & 0xFFFFu) << 48);   // tile_dim0 @bit112
    g1.q[2] = (unsigned long long)rows                        // tile_dim1 @bit128
            | ((unsigned long long)cols << 32);               // tensor_dim0_stride @bit160
    g1.q[3] = 0;                                              // stride hi / dim1_stride
    const i32x4 z4 = {0, 0, 0, 0};
    const i32x8 z8 = {0, 0, 0, 0, 0, 0, 0, 0};
    __builtin_amdgcn_tensor_load_to_lds(g0.v, g1.v, z4, z4, z8, 0);
}
#else
#define HAVE_TDM 0
#endif

// ---------------------------------------------------------------------------
__global__ void k_init(const float* __restrict__ W_enc, const float* __restrict__ W_ih,
                       const float* __restrict__ W_hh,  const float* __restrict__ W_dec,
                       const float* __restrict__ state,
                       bf16* wencB, bf16* wihB, bf16* whhB, bf16* wdecB, bf16* stateB,
                       float* hF, float* cF, bf16* hB, unsigned* barrier)
{
    const size_t i      = (size_t)blockIdx.x * blockDim.x + threadIdx.x;
    const size_t stride = (size_t)gridDim.x * blockDim.x;

    for (size_t k = i; k < (size_t)HID * OBSD; k += stride) wencB[k] = (bf16)W_enc[k];
    for (size_t k = i; k < (size_t)4 * HID * HID; k += stride) wihB[k] = (bf16)W_ih[k];
    for (size_t k = i; k < (size_t)4 * HID * HID; k += stride) whhB[k] = (bf16)W_hh[k];
    for (size_t k = i; k < (size_t)NPAD * HID; k += stride) {
        int row = (int)(k / HID);
        wdecB[k] = (row < ACT) ? (bf16)W_dec[k] : (bf16)0.0f;
    }
    for (size_t k = i; k < (size_t)T_STEPS * BATCH * OBSD; k += stride)
        stateB[k] = (bf16)state[k];
    for (size_t k = i; k < (size_t)BATCH * HID; k += stride) {
        hF[k] = 0.0f; cF[k] = 0.0f; hB[k] = (bf16)0.0f;
    }
    if (i == 0) *barrier = 0u;
}

// ---------------------------------------------------------------------------
// encoded = relu(state @ W_enc^T + b_enc), bf16 out.
// One wave computes a 16x64 slab (4 N-tiles); fragments double-buffered.
__global__ void k_encoder(const bf16* __restrict__ stateB, const bf16* __restrict__ wencB,
                          const float* __restrict__ b_enc, bf16* __restrict__ encB)
{
    const int wave = (int)((blockIdx.x * blockDim.x + threadIdx.x) >> 5);
    const int lane = threadIdx.x & 31;
    const int rowTile = wave >> 4;       // 2048 row tiles of 16 (M = T*B = 32768)
    const int nGroup  = wave & 15;       // 16 groups of 64 cols (H = 1024)
    const int row0 = rowTile * 16;
    const int col0 = nGroup * 64;
    const bf16* Abase = stateB + (size_t)row0 * OBSD;

    f32x8 acc[4] = {};
    bf16x16 a = ldfragA(Abase, OBSD);
    bf16x16 b[4];
#pragma unroll
    for (int n = 0; n < 4; ++n)
        b[n] = ldfragB(wencB + (size_t)(col0 + n * 16) * OBSD, OBSD);

    for (int k = 32; k <= OBSD; k += 32) {
        bf16x16 an, bn[4];
        if (k < OBSD) {
            an = ldfragA(Abase + k, OBSD);
#pragma unroll
            for (int n = 0; n < 4; ++n)
                bn[n] = ldfragB(wencB + (size_t)(col0 + n * 16) * OBSD + k, OBSD);
        }
#pragma unroll
        for (int n = 0; n < 4; ++n) acc[n] = wmma_bf16(a, b[n], acc[n]);
        if (k < OBSD) {
            a = an;
#pragma unroll
            for (int n = 0; n < 4; ++n) b[n] = bn[n];
        }
    }
#pragma unroll
    for (int n = 0; n < 4; ++n) {
        const int c = col0 + n * 16 + (lane & 15);
        const float bias = b_enc[c];
#pragma unroll
        for (int j = 0; j < 8; ++j) {
            const int r = row0 + j + ((lane >> 4) << 3);
            float v = acc[n][j] + bias;
            v = v > 0.0f ? v : 0.0f;
            encB[(size_t)r * HID + c] = (bf16)v;
        }
    }
}

// ---------------------------------------------------------------------------
// Persistent recurrent kernel: 32 blocks x 256 threads (8 wave32) = 256 waves.
// Block -> (rowGroup of 32 batch rows) x (8 h-col tiles, one per wave).
// Wave  -> 2 row-tiles x 4 gates = 8 acc tiles; each B fragment reused 2x.
__global__ __launch_bounds__(256, 1)
void k_recurrent(const bf16* __restrict__ encB, const bf16* __restrict__ wihB,
                 const bf16* __restrict__ whhB,
                 const float* __restrict__ b_ih, const float* __restrict__ b_hh,
                 const int* __restrict__ mask,
                 bf16* hB, float* hF, float* cF, bf16* latB, unsigned* barrier)
{
    extern __shared__ __align__(16) bf16 smem[];   // 2 x 32x1024 bf16 = 128 KB
    bf16* sEnc = smem;                 // enc_t rows [row0, row0+32)
    bf16* sH   = smem + 32 * HID;      // h     rows [row0, row0+32)

    const int tid   = threadIdx.x;
    const int lane  = tid & 31;
    const int waveB = tid >> 5;                     // 0..7
    const int rowGroup = blockIdx.x >> 3;           // 0..3  (128 batch rows / 32)
    const int colBlk   = blockIdx.x & 7;            // 0..7
    const int cg       = colBlk * 8 + waveB;        // 0..63 (1024 h-cols / 16)
    const int row0     = rowGroup * 32;
    const int col16    = cg * 16;
    const int c        = col16 + (lane & 15);       // this lane's h column

    const float bi_i = b_ih[0 * HID + c] + b_hh[0 * HID + c];
    const float bi_f = b_ih[1 * HID + c] + b_hh[1 * HID + c];
    const float bi_g = b_ih[2 * HID + c] + b_hh[2 * HID + c];
    const float bi_o = b_ih[3 * HID + c] + b_hh[3 * HID + c];

    for (int t = 0; t < T_STEPS; ++t) {
        // ---- stage A slabs (32x1024 bf16 each) into LDS ----
        const bf16* encSrc = encB + ((size_t)t * BATCH + row0) * HID;
        const bf16* hSrc   = hB + (size_t)row0 * HID;
#if HAVE_TDM
        if (waveB == 0) {               // one wave drives the Tensor Data Mover
            tdm_load_2d(encSrc, sEnc, 32, HID);
            tdm_load_2d(hSrc,   sH,   32, HID);
            __builtin_amdgcn_s_wait_tensorcnt(0);
        }
        __syncthreads();
#else
        {
            const bf16x8* eS = (const bf16x8*)encSrc;
            const bf16x8* hS = (const bf16x8*)hSrc;
            bf16x8* dE = (bf16x8*)sEnc;
            bf16x8* dH = (bf16x8*)sH;
            for (int idx = tid; idx < (32 * HID) / 8; idx += 256) {
                dE[idx] = eS[idx];
                dH[idx] = hS[idx];
            }
        }
        __syncthreads();
#endif

        // ---- gates = enc_t @ W_ih^T + h @ W_hh^T  (K = 1024 + 1024) ----
        f32x8 acc[2][4] = {};
#pragma unroll
        for (int part = 0; part < 2; ++part) {
            const bf16* As = part ? sH : sEnc;
            const bf16* W  = part ? whhB : wihB;

            bf16x16 a0 = ldfragA(As, HID);
            bf16x16 a1 = ldfragA(As + 16 * HID, HID);
            bf16x16 b[4];
#pragma unroll
            for (int g = 0; g < 4; ++g)
                b[g] = ldfragB(W + (size_t)(g * HID + col16) * HID, HID);

            for (int k = 32; k <= HID; k += 32) {
                bf16x16 a0n, a1n, bn[4];
                if (k < HID) {
                    __builtin_prefetch(W + (size_t)col16 * HID + k + 64, 0, 0);
                    a0n = ldfragA(As + k, HID);
                    a1n = ldfragA(As + 16 * HID + k, HID);
#pragma unroll
                    for (int g = 0; g < 4; ++g)
                        bn[g] = ldfragB(W + (size_t)(g * HID + col16) * HID + k, HID);
                }
#pragma unroll
                for (int g = 0; g < 4; ++g) {
                    acc[0][g] = wmma_bf16(a0, b[g], acc[0][g]);
                    acc[1][g] = wmma_bf16(a1, b[g], acc[1][g]);
                }
                if (k < HID) {
                    a0 = a0n; a1 = a1n;
#pragma unroll
                    for (int g = 0; g < 4; ++g) b[g] = bn[g];
                }
            }
        }

        // ---- LSTM cell update (lane-local: col c, 2x8 rows per lane) ----
#pragma unroll
        for (int mt = 0; mt < 2; ++mt) {
#pragma unroll
            for (int j = 0; j < 8; ++j) {
                const int r = row0 + mt * 16 + j + ((lane >> 4) << 3);
                const float m  = (float)mask[t * BATCH + r];
                const float gi = sigmoidf_(acc[mt][0][j] + bi_i);
                const float gf = sigmoidf_(acc[mt][1][j] + bi_f);
                const float gg = tanhf    (acc[mt][2][j] + bi_g);
                const float go = sigmoidf_(acc[mt][3][j] + bi_o);
                const size_t idx = (size_t)r * HID + c;
                const float cn = gf * cF[idx] + gi * gg;
                float hn = go * tanhf(cn);
                cF[idx] = cn * m;
                hn *= m;
                hF[idx] = hn;
                hB[idx] = (bf16)hn;
                latB[((size_t)t * BATCH + r) * HID + c] = (bf16)hn;
            }
        }

        // ---- grid-wide barrier before next step consumes new h ----
        __threadfence();
        __syncthreads();
        if (tid == 0) {
            atomicAdd(barrier, 1u);
            const unsigned expected = (unsigned)gridDim.x * (unsigned)(t + 1);
            while (__hip_atomic_load(barrier, __ATOMIC_RELAXED,
                                     __HIP_MEMORY_SCOPE_AGENT) < expected) {
                __builtin_amdgcn_s_sleep(1);
            }
        }
        __syncthreads();
        __threadfence();
    }
}

// ---------------------------------------------------------------------------
// values = (latent @ W_dec^T + b_dec) * mask ; N padded 18 -> 32 (2 tiles).
__global__ void k_decoder(const bf16* __restrict__ latB, const bf16* __restrict__ wdecB,
                          const float* __restrict__ b_dec, const int* __restrict__ mask,
                          float* __restrict__ out)
{
    const int wave = (int)((blockIdx.x * blockDim.x + threadIdx.x) >> 5); // 0..2047
    const int lane = threadIdx.x & 31;
    const int row0 = wave * 16;
    const bf16* Abase = latB + (size_t)row0 * HID;

    f32x8 acc[2] = {};
    bf16x16 a = ldfragA(Abase, HID);
    bf16x16 b[2];
#pragma unroll
    for (int n = 0; n < 2; ++n) b[n] = ldfragB(wdecB + (size_t)(n * 16) * HID, HID);

    for (int k = 32; k <= HID; k += 32) {
        bf16x16 an, bn[2];
        if (k < HID) {
            an = ldfragA(Abase + k, HID);
#pragma unroll
            for (int n = 0; n < 2; ++n)
                bn[n] = ldfragB(wdecB + (size_t)(n * 16) * HID + k, HID);
        }
#pragma unroll
        for (int n = 0; n < 2; ++n) acc[n] = wmma_bf16(a, b[n], acc[n]);
        if (k < HID) {
            a = an;
#pragma unroll
            for (int n = 0; n < 2; ++n) b[n] = bn[n];
        }
    }
#pragma unroll
    for (int n = 0; n < 2; ++n) {
        const int col = n * 16 + (lane & 15);
        if (col < ACT) {
            const float bias = b_dec[col];
#pragma unroll
            for (int j = 0; j < 8; ++j) {
                const int r = row0 + j + ((lane >> 4) << 3);
                const float m = (float)mask[r];
                out[(size_t)r * ACT + col] = (acc[n][j] + bias) * m;
            }
        }
    }
}

// ---------------------------------------------------------------------------
__global__ void k_finalize(const float* __restrict__ hF, const float* __restrict__ cF,
                           float* __restrict__ out)
{
    const int i = blockIdx.x * blockDim.x + threadIdx.x;
    if (i < BATCH * HID) {
        const size_t base = (size_t)T_STEPS * BATCH * ACT;
        out[base + i]               = hF[i];
        out[base + BATCH * HID + i] = cF[i];
    }
}

// ---------------------------------------------------------------------------
extern "C" void kernel_launch(void* const* d_in, const int* in_sizes, int n_in,
                              void* d_out, int out_size, void* d_ws, size_t ws_size,
                              hipStream_t stream)
{
    const float* state = (const float*)d_in[0];
    const int*   mask  = (const int*)  d_in[1];
    const float* W_enc = (const float*)d_in[2];
    const float* b_enc = (const float*)d_in[3];
    const float* W_ih  = (const float*)d_in[4];
    const float* W_hh  = (const float*)d_in[5];
    const float* b_ih  = (const float*)d_in[6];
    const float* b_hh  = (const float*)d_in[7];
    const float* W_dec = (const float*)d_in[8];
    const float* b_dec = (const float*)d_in[9];
    float* out = (float*)d_out;

    char* ws = (char*)d_ws;
    size_t o = 0;
    auto take = [&](size_t bytes) -> char* {
        char* p = ws + o;
        o += (bytes + 255) & ~(size_t)255;
        return p;
    };
    bf16*     stateB  = (bf16*)    take((size_t)T_STEPS * BATCH * OBSD * sizeof(bf16));
    bf16*     wencB   = (bf16*)    take((size_t)HID * OBSD * sizeof(bf16));
    bf16*     wihB    = (bf16*)    take((size_t)4 * HID * HID * sizeof(bf16));
    bf16*     whhB    = (bf16*)    take((size_t)4 * HID * HID * sizeof(bf16));
    bf16*     wdecB   = (bf16*)    take((size_t)NPAD * HID * sizeof(bf16));
    bf16*     encB    = (bf16*)    take((size_t)T_STEPS * BATCH * HID * sizeof(bf16));
    bf16*     latB    = (bf16*)    take((size_t)T_STEPS * BATCH * HID * sizeof(bf16));
    bf16*     hB      = (bf16*)    take((size_t)BATCH * HID * sizeof(bf16));
    float*    hF      = (float*)   take((size_t)BATCH * HID * sizeof(float));
    float*    cF      = (float*)   take((size_t)BATCH * HID * sizeof(float));
    unsigned* barrier = (unsigned*)take(256);

    k_init<<<2048, 256, 0, stream>>>(W_enc, W_ih, W_hh, W_dec, state,
                                     wencB, wihB, whhB, wdecB, stateB,
                                     hF, cF, hB, barrier);
    // 32768 waves -> 4096 blocks of 8 waves
    k_encoder<<<4096, 256, 0, stream>>>(stateB, wencB, b_enc, encB);
    // persistent: 32 co-resident blocks, 128 KB dynamic LDS each (2/WGP)
    k_recurrent<<<32, 256, 2 * 32 * HID * sizeof(bf16), stream>>>(
        encB, wihB, whhB, b_ih, b_hh, mask, hB, hF, cF, latB, barrier);
    // 2048 waves -> 256 blocks
    k_decoder<<<256, 256, 0, stream>>>(latB, wdecB, b_dec, mask, out);
    k_finalize<<<512, 256, 0, stream>>>(hF, cF, out);
}
